// ScatterMoE_31920196944055
// MI455X (gfx1250) — matrix operand order repaired and checked by reading, not compiled, not run on hardware
//
#include <hip/hip_runtime.h>

// ---------------------------------------------------------------------------
// MoE (top-2) fused MLP for gfx1250 (CDNA5, wave32, WMMA bf16 16x16x32).
//   zero_out      : zero d_out and per-expert counters
//   router_kernel : logits = x @ Wr^T, top-2 softmax, per-expert token lists
//   moe_expert    : per (expert, 16-token tile), per 128-wide H block:
//                     htile = relu(x_tile @ W1[:,hb:hb+128])^2   (bf16 WMMA)
//                     out_acc += htile @ W2[hb:hb+128,:]         (bf16 WMMA)
//                   epilogue: atomicAdd(gate * out_acc) into d_out
// ---------------------------------------------------------------------------

typedef __bf16          bf16x16 __attribute__((ext_vector_type(16)));
typedef float           f32x8   __attribute__((ext_vector_type(8)));
typedef float           f32x4   __attribute__((ext_vector_type(4)));
typedef unsigned short  u16x8   __attribute__((ext_vector_type(8)));
typedef unsigned short  u16x4   __attribute__((ext_vector_type(4)));

union FragBF {
    struct { u16x8 lo, hi; } p;   // two 16-byte chunks per lane
    bf16x16 v;
};

// native RNE convert: lowers to v_cvt_pk_bf16_f32 on gfx1250
__device__ __forceinline__ unsigned short bfbits(float f) {
    union { __bf16 h; unsigned short u; } c;
    c.h = (__bf16)f;
    return c.u;
}

#define C_DIM   1024
#define H_DIM   4096
#define E_NUM   8
#define TILE_M  16
#define HB      128          // H-block width per outer iteration
#define KC_UP   64           // K staged per up-proj step (2 WMMA k-steps)

// LDS strides in halves (fragment chunks stay 16B-aligned)
#define XS_STRIDE   1032     // x tile rows       (1024 + 8 pad)
#define W1T_STRIDE  72       // W1 chunk, K-major (64 + 8 pad)
#define HS_STRIDE   136      // htile rows        (128 + 8 pad)
#define W2T_STRIDE  40       // W2 chunk, K-major (32 + 8 pad)

#define XS_HALVES   (TILE_M * XS_STRIDE)    // 16512
#define W1T_HALVES  (HB * W1T_STRIDE)       //  9216
#define HS_HALVES   (TILE_M * HS_STRIDE)    //  2176
#define W2T_HALVES  (C_DIM * W2T_STRIDE)    // 40960
#define SMEM_BYTES  ((XS_HALVES + W1T_HALVES + HS_HALVES + W2T_HALVES) * 2 \
                     + TILE_M * 4 + TILE_M * 4)

// ---------------------------------------------------------------------------
__global__ __launch_bounds__(256) void zero_out(float* __restrict__ out,
                                                int n_out,
                                                int* __restrict__ counts) {
    int i = blockIdx.x * blockDim.x + threadIdx.x;
    if (i < n_out) out[i] = 0.0f;
    if (i < E_NUM) counts[i] = 0;
}

// ---------------------------------------------------------------------------
__global__ __launch_bounds__(256) void router_kernel(const float* __restrict__ x,
                                                     const float* __restrict__ Wr,
                                                     int* __restrict__ counts,
                                                     int* __restrict__ rows,
                                                     float* __restrict__ wts,
                                                     int N) {
    int t = blockIdx.x * blockDim.x + threadIdx.x;
    if (t >= N) return;

    const float* xr = x + (size_t)t * C_DIM;
    float acc[E_NUM];
#pragma unroll
    for (int e = 0; e < E_NUM; ++e) acc[e] = 0.0f;

    for (int c = 0; c < C_DIM; c += 4) {
        f32x4 xv = *(const f32x4*)(xr + c);
#pragma unroll
        for (int e = 0; e < E_NUM; ++e) {
            f32x4 wv = *(const f32x4*)(Wr + e * C_DIM + c);
            float s = acc[e];
            s = fmaf(xv[0], wv[0], s);
            s = fmaf(xv[1], wv[1], s);
            s = fmaf(xv[2], wv[2], s);
            s = fmaf(xv[3], wv[3], s);
            acc[e] = s;
        }
    }

    // top-2 (strict > keeps lowest index on ties, matching lax.top_k)
    int e0 = 0;
#pragma unroll
    for (int e = 1; e < E_NUM; ++e) if (acc[e] > acc[e0]) e0 = e;
    int e1 = (e0 == 0) ? 1 : 0;
#pragma unroll
    for (int e = 0; e < E_NUM; ++e)
        if (e != e0 && acc[e] > acc[e1]) e1 = e;

    float ex = __expf(acc[e1] - acc[e0]);     // <= 1
    float w0 = 1.0f / (1.0f + ex);
    float w1 = ex * w0;

    int p0 = atomicAdd(&counts[e0], 1);
    rows[e0 * N + p0] = t;  wts[e0 * N + p0] = w0;
    int p1 = atomicAdd(&counts[e1], 1);
    rows[e1 * N + p1] = t;  wts[e1 * N + p1] = w1;
}

// ---------------------------------------------------------------------------
__global__ __launch_bounds__(256) void moe_expert_kernel(
        const float* __restrict__ x,
        const float* __restrict__ W1,
        const float* __restrict__ W2,
        float* __restrict__ out,
        const int* __restrict__ rows,
        const float* __restrict__ wts,
        const int* __restrict__ counts,
        int N) {
    extern __shared__ unsigned short smem[];
    unsigned short* xs   = smem;                     // bf16 x tile   [16][1032]
    unsigned short* w1t  = xs  + XS_HALVES;          // bf16 W1 chunk [128][72] (k-major)
    unsigned short* hs   = w1t + W1T_HALVES;         // bf16 htile    [16][136]
    unsigned short* w2t  = hs  + HS_HALVES;          // bf16 W2 chunk [1024][40] (k-major)
    int*   tok_s = (int*)(w2t + W2T_HALVES);
    float* twt_s = (float*)(tok_s + TILE_M);

    const int e   = blockIdx.y;
    const int rb  = blockIdx.x;
    const int cnt = counts[e];
    const int m0  = rb * TILE_M;
    if (m0 >= cnt) return;                            // block-uniform exit

    const int tid   = threadIdx.x;
    const int wave  = tid >> 5;
    const int lane  = tid & 31;
    const int lan15 = lane & 15;
    const int hsel  = lane >> 4;                      // 0|1 : lane half (K split)

    // ---- token ids + gate weights for this 16-row tile (pad weight = 0) ----
    if (tid < TILE_M) {
        int r = m0 + tid;
        bool valid = (r < cnt);
        tok_s[tid] = rows[e * N + (valid ? r : m0)];
        twt_s[tid] = valid ? wts[e * N + r] : 0.0f;
    }
    __syncthreads();

    // ---- stage x tile as bf16 (saddr-form float4 loads, b64 LDS stores) ----
#pragma unroll
    for (int r = 0; r < TILE_M; ++r) {
        int tok = __builtin_amdgcn_readfirstlane(tok_s[r]);   // uniform row base
        const float* xrow = x + (size_t)tok * C_DIM;
        int c0 = tid << 2;
        f32x4 v = *(const f32x4*)(xrow + c0);
        u16x4 h;
#pragma unroll
        for (int j = 0; j < 4; ++j) h[j] = bfbits(v[j]);
        *(u16x4*)(xs + r * XS_STRIDE + c0) = h;
    }

    const float* W1e = W1 + (size_t)e * C_DIM * H_DIM;
    const float* W2e = W2 + (size_t)e * H_DIM * C_DIM;

    f32x8 oacc[8];                                    // wave's 16 x 128 out slice
    {
        f32x8 z = {};
#pragma unroll
        for (int ct = 0; ct < 8; ++ct) oacc[ct] = z;
    }

    for (int hb = 0; hb < H_DIM; hb += HB) {
        // ================= up-projection: htile[16][128] ====================
        f32x8 hacc = {};
        for (int kc = 0; kc < C_DIM; kc += KC_UP) {
            __syncthreads();                          // previous w1t readers done
            // stage W1[kc:kc+64, hb:hb+128] transposed: 4x4 micro-tiles
            {
                const float* w1base = W1e + (size_t)kc * H_DIM + hb;  // uniform
#pragma unroll
                for (int it = 0; it < 2; ++it) {
                    int id = tid + (it << 8);
                    int k0 = (id >> 5) << 2;          // 0..60 step 4
                    int h0 = (id & 31) << 2;          // 0..124 step 4
                    f32x4 row[4];
#pragma unroll
                    for (int j = 0; j < 4; ++j)
                        row[j] = *(const f32x4*)(w1base + (k0 + j) * H_DIM + h0);
#pragma unroll
                    for (int i2 = 0; i2 < 4; ++i2) {
                        u16x4 hh;
#pragma unroll
                        for (int j = 0; j < 4; ++j) hh[j] = bfbits(row[j][i2]);
                        *(u16x4*)(w1t + (h0 + i2) * W1T_STRIDE + k0) = hh;
                    }
                }
            }
            __syncthreads();

#pragma unroll
            for (int kk = 0; kk < KC_UP; kk += 32) {
                FragBF a, b;
                const unsigned short* ap = xs + lan15 * XS_STRIDE + (kc + kk) + hsel * 16;
                a.p.lo = *(const u16x8*)(ap);
                a.p.hi = *(const u16x8*)(ap + 8);
                const unsigned short* bp =
                    w1t + (wave * 16 + lan15) * W1T_STRIDE + kk + hsel * 16;
                b.p.lo = *(const u16x8*)(bp);
                b.p.hi = *(const u16x8*)(bp + 8);
                hacc = __builtin_amdgcn_wmma_f32_16x16x32_bf16(
                           false, a.v, false, b.v, (short)0, hacc, false, false);
            }
        }

        // relu^2 -> bf16 htile in LDS (D layout: m = hsel*8+v, n = lane&15)
        {
            int n  = wave * 16 + lan15;
            int mb = hsel * 8;
#pragma unroll
            for (int v = 0; v < 8; ++v) {
                float f = fmaxf(hacc[v], 0.0f);
                hs[(mb + v) * HS_STRIDE + n] = bfbits(f * f);
            }
        }

        // ================= down-projection: out += htile @ W2 ===============
        for (int ks = 0; ks < HB; ks += 32) {
            __syncthreads();                          // w2t readers + hs writers done
            // stage W2[hb+ks : +32, 0:1024] transposed: 4x4 micro-tiles
            {
                const float* w2base = W2e + (size_t)(hb + ks) * C_DIM;  // uniform
#pragma unroll
                for (int it = 0; it < 8; ++it) {
                    int id = tid + (it << 8);         // 0..2047
                    int k0 = (id >> 8) << 2;          // 0..28 step 4
                    int c0 = (id & 255) << 2;         // 0..1020 step 4
                    f32x4 row[4];
#pragma unroll
                    for (int j = 0; j < 4; ++j)
                        row[j] = *(const f32x4*)(w2base + (k0 + j) * C_DIM + c0);
#pragma unroll
                    for (int i2 = 0; i2 < 4; ++i2) {
                        u16x4 hh;
#pragma unroll
                        for (int j = 0; j < 4; ++j) hh[j] = bfbits(row[j][i2]);
                        *(u16x4*)(w2t + (c0 + i2) * W2T_STRIDE + k0) = hh;
                    }
                }
            }
            __syncthreads();

            FragBF a;
            const unsigned short* ap = hs + lan15 * HS_STRIDE + ks + hsel * 16;
            a.p.lo = *(const u16x8*)(ap);
            a.p.hi = *(const u16x8*)(ap + 8);
#pragma unroll
            for (int ct = 0; ct < 8; ++ct) {
                FragBF b;
                const unsigned short* bp =
                    w2t + (wave * 128 + ct * 16 + lan15) * W2T_STRIDE + hsel * 16;
                b.p.lo = *(const u16x8*)(bp);
                b.p.hi = *(const u16x8*)(bp + 8);
                oacc[ct] = __builtin_amdgcn_wmma_f32_16x16x32_bf16(
                               false, a.v, false, b.v, (short)0, oacc[ct], false, false);
            }
        }
    }

    // ---- epilogue: gate-weighted atomic scatter-add into out ---------------
    // 32-bit offsets (max N*C*4 = 33 MB) -> saddr-form global atomics
    {
        int mb = hsel * 8;
#pragma unroll
        for (int ct = 0; ct < 8; ++ct) {
            int cc = wave * 128 + ct * 16 + lan15;
#pragma unroll
            for (int v = 0; v < 8; ++v) {
                int m   = mb + v;
                int off = tok_s[m] * C_DIM + cc;
                atomicAdd(out + off, twt_s[m] * oacc[ct][v]);
            }
        }
    }
}

// ---------------------------------------------------------------------------
extern "C" void kernel_launch(void* const* d_in, const int* in_sizes, int n_in,
                              void* d_out, int out_size, void* d_ws, size_t ws_size,
                              hipStream_t stream) {
    const float* x  = (const float*)d_in[0];
    const float* Wr = (const float*)d_in[1];
    const float* W1 = (const float*)d_in[2];
    const float* W2 = (const float*)d_in[3];
    float* out = (float*)d_out;

    const int N = in_sizes[0] / C_DIM;               // B*T = 8192

    // workspace layout
    int*   counts = (int*)d_ws;                      // [16] (8 used, padded)
    int*   rows   = counts + 16;                     // [E][N]
    float* wts    = (float*)(rows + E_NUM * N);      // [E][N]

    const int n_out = N * C_DIM;
    zero_out<<<(n_out + 255) / 256, 256, 0, stream>>>(out, n_out, counts);
    router_kernel<<<(N + 255) / 256, 256, 0, stream>>>(x, Wr, counts, rows, wts, N);

    dim3 grid((N + TILE_M - 1) / TILE_M, E_NUM);     // (512, 8)
    moe_expert_kernel<<<grid, 256, SMEM_BYTES, stream>>>(
        x, W1, W2, out, rows, wts, counts, N);
}
